// TopKRouter_22265110463277
// MI455X (gfx1250) — compile-verified, matrix operand order checked by reading, not run on hardware
//
#include <hip/hip_runtime.h>

// ---- types matching the gfx1250 WMMA builtin signatures ----
typedef __attribute__((ext_vector_type(16))) __bf16 v16bf;
typedef __attribute__((ext_vector_type(8)))  __bf16 v8bf;
typedef __attribute__((ext_vector_type(8)))  float  v8f;
typedef __attribute__((ext_vector_type(4)))  float  v4f;

#define D_MODEL     4096
#define NUM_EXPERTS 64
#define TOKENS      16384          // 4 * 4096
#define TOKW        16             // tokens per wave (WMMA M)
#define KSPLIT      2              // K split across wave pairs
#define KHALF       (D_MODEL / KSPLIT)

// ---------------------------------------------------------------------------
// Prologue: split f32 gate weights into bf16 hi/lo once (W is 1 MB, L2-hot).
// ---------------------------------------------------------------------------
__global__ void wsplit_kernel(const float* __restrict__ W,
                              __bf16* __restrict__ whi,
                              __bf16* __restrict__ wlo, int n) {
  int i = blockIdx.x * blockDim.x + threadIdx.x;
  if (i < n) {
    float w = W[i];
    __bf16 h = (__bf16)w;
    whi[i] = h;
    wlo[i] = (__bf16)(w - (float)h);
  }
}

// ---------------------------------------------------------------------------
// Main: [16384,4096] x [4096,64] gate GEMM via compensated bf16 WMMA,
// then per-token softmax + top-2 scatter.
// Block = 128 threads = 4 waves; waves (0,1) and (2,3) pair up on one
// 16-token tile, each doing half of K; partials merged in LDS.
// ---------------------------------------------------------------------------
__global__ __launch_bounds__(128) void router_kernel(
    const float*  __restrict__ x,
    const __bf16* __restrict__ whi,
    const __bf16* __restrict__ wlo,
    const float*  __restrict__ bias,
    float*        __restrict__ out) {
  // 2 token tiles per block, 16 x (64+4) padded rows (stride 68 -> banks t*4)
  __shared__ float lds[2][TOKW][NUM_EXPERTS + 4];

  const int tid  = threadIdx.x;
  const int wave = tid >> 5;
  const int lane = tid & 31;
  const int half = lane >> 4;     // WMMA lane-half
  const int r    = lane & 15;
  const int pair = wave >> 1;     // which 16-token tile in this block
  const int ksel = wave & 1;      // which K half this wave computes
  const int m0   = (blockIdx.x * 2 + pair) * TOKW;

  // A: row (m0+r), K segments {8h..8h+7} and {16+8h..16+8h+7} per k0-step
  const float* px = x + (size_t)(m0 + r) * D_MODEL + ksel * KHALF + half * 8;
  // B: expert row (16t + r), same K segment pattern
  const size_t wofs = (size_t)r * D_MODEL + ksel * KHALF + half * 8;

  v8f acc[4];
#pragma unroll
  for (int t = 0; t < 4; ++t) acc[t] = (v8f){};

#pragma unroll 2
  for (int k0 = 0; k0 < KHALF; k0 += 32) {
    if (k0 + 288 <= KHALF)
      __builtin_prefetch(px + k0 + 256, 0, 0);   // global_prefetch_b8 on x stream

    // ---- A fragment: 16 f32 -> bf16 hi/lo ----
    v4f f0 = *(const v4f*)(px + k0);
    v4f f1 = *(const v4f*)(px + k0 + 4);
    v4f f2 = *(const v4f*)(px + k0 + 16);
    v4f f3 = *(const v4f*)(px + k0 + 20);
    float fa[16];
#pragma unroll
    for (int i = 0; i < 4; ++i) {
      fa[i] = f0[i]; fa[4 + i] = f1[i]; fa[8 + i] = f2[i]; fa[12 + i] = f3[i];
    }
    v16bf ahi, alo;
#pragma unroll
    for (int i = 0; i < 16; ++i) {
      __bf16 h = (__bf16)fa[i];
      ahi[i] = h;
      alo[i] = (__bf16)(fa[i] - (float)h);
    }

    // ---- 4 expert tiles: B hi/lo loads + 3 compensated WMMAs each ----
#pragma unroll
    for (int t = 0; t < 4; ++t) {
      const __bf16* wp = whi + (size_t)(16 * t) * D_MODEL + wofs + k0;
      const __bf16* wq = wlo + (size_t)(16 * t) * D_MODEL + wofs + k0;
      v8bf h0 = *(const v8bf*)(wp);
      v8bf h1 = *(const v8bf*)(wp + 16);
      v8bf l0 = *(const v8bf*)(wq);
      v8bf l1 = *(const v8bf*)(wq + 16);
      v16bf bh, bl;
#pragma unroll
      for (int i = 0; i < 8; ++i) {
        bh[i] = h0[i]; bh[8 + i] = h1[i];
        bl[i] = l0[i]; bl[8 + i] = l1[i];
      }
      acc[t] = __builtin_amdgcn_wmma_f32_16x16x32_bf16(
          false, ahi, false, bh, (short)0, acc[t], false, false);
      acc[t] = __builtin_amdgcn_wmma_f32_16x16x32_bf16(
          false, ahi, false, bl, (short)0, acc[t], false, false);
      acc[t] = __builtin_amdgcn_wmma_f32_16x16x32_bf16(
          false, alo, false, bh, (short)0, acc[t], false, false);
    }
  }

  // ---- merge K-split partials in LDS ----
  // C layout: VGPR v, lanes 0-15 -> M=v, lanes 16-31 -> M=v+8; N = 16t + r
  if (ksel == 0) {
#pragma unroll
    for (int t = 0; t < 4; ++t)
#pragma unroll
      for (int v = 0; v < 8; ++v)
        lds[pair][half * 8 + v][t * 16 + r] = acc[t][v];
  }
  __syncthreads();
  if (ksel == 1) {
#pragma unroll
    for (int t = 0; t < 4; ++t)
#pragma unroll
      for (int v = 0; v < 8; ++v)
        lds[pair][half * 8 + v][t * 16 + r] += acc[t][v];
  }
  __syncthreads();

  // ---- softmax + top-2 scatter: lane t of the even wave owns token t ----
  if (ksel == 0 && lane < 16) {
    const int token = m0 + lane;
    float l[NUM_EXPERTS];
    float m = -INFINITY;
#pragma unroll
    for (int e = 0; e < NUM_EXPERTS; ++e) {
      float v = lds[pair][lane][e] + bias[e];
      l[e] = v;
      m = fmaxf(m, v);
    }
    float s = 0.0f;
    float v1 = -INFINITY, v2 = -INFINITY;
    int   i1 = -1, i2 = -1;
#pragma unroll
    for (int e = 0; e < NUM_EXPERTS; ++e) {
      s += __expf(l[e] - m);
      if (l[e] > v1)      { v2 = v1; i2 = i1; v1 = l[e]; i1 = e; }
      else if (l[e] > v2) { v2 = l[e]; i2 = e; }
    }
    const float inv = 1.0f / s;
    float* po = out + (size_t)token * NUM_EXPERTS;
#pragma unroll
    for (int e = 0; e < NUM_EXPERTS; e += 4) {
      v4f o;
#pragma unroll
      for (int j = 0; j < 4; ++j) {
        int ee = e + j;
        o[j] = (ee == i1 || ee == i2) ? __expf(l[ee] - m) * inv : 0.0f;
      }
      *(v4f*)(po + e) = o;
    }
  }
}

// ---------------------------------------------------------------------------
extern "C" void kernel_launch(void* const* d_in, const int* in_sizes, int n_in,
                              void* d_out, int out_size, void* d_ws, size_t ws_size,
                              hipStream_t stream) {
  const float* x = (const float*)d_in[0];   // [4,4096,4096] f32
  const float* W = (const float*)d_in[1];   // [64,4096] f32
  const float* b = (const float*)d_in[2];   // [64] f32
  float* out = (float*)d_out;               // [4,4096,64] f32

  __bf16* whi = (__bf16*)d_ws;                                  // 512 KB
  __bf16* wlo = whi + (size_t)NUM_EXPERTS * D_MODEL;            // 512 KB
  const int nw = NUM_EXPERTS * D_MODEL;                         // 262144

  wsplit_kernel<<<(nw + 255) / 256, 256, 0, stream>>>(W, whi, wlo, nw);

  // 512 blocks x 4 waves: 2048 waves in flight, each streams 128 KB of x
  router_kernel<<<TOKENS / (2 * TOKW), 128, 0, stream>>>(x, whi, wlo, b, out);
}
